// Deform_Transformer_33509334843487
// MI455X (gfx1250) — matrix operand order repaired and checked by reading, not compiled, not run on hardware
//
#include <hip/hip_runtime.h>
#include <math.h>

// ---------------- problem constants ----------------
#define NB     4
#define LQ     1000
#define DIMC   256
#define HEADS  8
#define DHEAD  32
#define LVLS   5
#define NPTS   4
#define MLPD   512
#define LINR   21824
#define QROWS  (NB * LQ)     // 4000
#define SROWS  (NB * LINR)   // 87296
#define ATT_SCALE 0.17677669529663687f  // 32^-0.5

typedef __attribute__((ext_vector_type(16))) _Float16 v16h;
typedef __attribute__((ext_vector_type(8)))  _Float16 v8h;
typedef __attribute__((ext_vector_type(4)))  _Float16 v4h;
typedef __attribute__((ext_vector_type(8)))  float    v8f;

#if defined(__has_builtin)
#if __has_builtin(__builtin_amdgcn_tensor_load_to_lds) && __has_builtin(__builtin_amdgcn_s_wait_tensorcnt)
#define USE_TDM 1
#endif
#endif

#ifdef USE_TDM
typedef unsigned int v4u __attribute__((ext_vector_type(4)));
typedef int v4i __attribute__((ext_vector_type(4)));
typedef int v8i __attribute__((ext_vector_type(8)));

// TDM: load a 64x32-element (rows x K) f16 tile from global Bt[N][K] into LDS,
// LDS rows padded to 40 halfs (64B data + 16B pad) via pad_interval/amount.
// Rows >= rem_rows are zero-filled by tensor OOB handling.
static __device__ __forceinline__ void tdm_load_tileB(const _Float16* gsrc,
                                                      unsigned lds_off,
                                                      int rem_rows, int K) {
  unsigned long long ga = (unsigned long long)(uintptr_t)gsrc;
  v4u g0;
  g0[0] = 1u;                                                   // count=1 (valid user D#)
  g0[1] = lds_off;                                              // LDS byte address
  g0[2] = (unsigned)ga;                                         // global_addr[31:0]
  g0[3] = (unsigned)((ga >> 32) & 0x01FFFFFFull) | (2u << 30);  // [56:32] | type=2
  v8i g1;
  g1[0] = (1 << 16)   // data_size = 2 bytes
        | (1 << 20)   // pad_enable
        | (3 << 22)   // pad_interval: 16 DWORDs (= one 64B tile row)
        | (3 << 25);  // pad_amount: 4 DWORDs (= 16B -> 40-half row stride)
  g1[1] = (int)((K & 0xFFFF) << 16);                            // tensor_dim0 lo16
  g1[2] = (int)(((unsigned)K >> 16) & 0xFFFF)                   // tensor_dim0 hi16
        | (int)((rem_rows & 0xFFFF) << 16);                     // tensor_dim1 lo16
  g1[3] = (32 << 16);                                           // tdim1 hi=0 | tile_dim0=32
  g1[4] = 64;                                                   // tile_dim1=64, tile_dim2=0
  g1[5] = (int)K;                                               // tensor_dim0_stride lo32
  g1[6] = 0;
  g1[7] = 0;
  v4i z4i; z4i[0] = 0; z4i[1] = 0; z4i[2] = 0; z4i[3] = 0;
  v8i z8i;
#pragma unroll
  for (int j = 0; j < 8; ++j) z8i[j] = 0;
  // 6-arg form (clang-23 / therock headers): groups 2,3 + extra group, cpol
  __builtin_amdgcn_tensor_load_to_lds(g0, g1, z4i, z4i, z8i, 0);
}
#endif

static __device__ __forceinline__ v16h mk16(v8h lo, v8h hi) {
  v16h r;
#pragma unroll
  for (int j = 0; j < 8; ++j) { r[j] = lo[j]; r[j + 8] = hi[j]; }
  return r;
}

static __device__ __forceinline__ v8h cvt8(float4 f0, float4 f1) {
  v8h h;
  h[0] = (_Float16)f0.x; h[1] = (_Float16)f0.y; h[2] = (_Float16)f0.z; h[3] = (_Float16)f0.w;
  h[4] = (_Float16)f1.x; h[5] = (_Float16)f1.y; h[6] = (_Float16)f1.z; h[7] = (_Float16)f1.w;
  return h;
}

static __device__ __forceinline__ v8h cvt8s(float4 f0, float4 f1, float s) {
  v8h h;
  h[0] = (_Float16)(f0.x * s); h[1] = (_Float16)(f0.y * s);
  h[2] = (_Float16)(f0.z * s); h[3] = (_Float16)(f0.w * s);
  h[4] = (_Float16)(f1.x * s); h[5] = (_Float16)(f1.y * s);
  h[6] = (_Float16)(f1.z * s); h[7] = (_Float16)(f1.w * s);
  return h;
}

static __device__ __forceinline__ float gelu_f(float x) {
  const float c = 0.7978845608028654f;  // sqrt(2/pi)
  return 0.5f * x * (1.0f + tanhf(c * (x + 0.044715f * x * x * x)));
}

// ---------------- weight transpose+convert: W[K][Nc] f32 -> Bt[Nc][K] f16 ----------------
__global__ void wtrans(const float* __restrict__ W, _Float16* __restrict__ Bt,
                       int K, int Nc) {
  int k = blockIdx.x * 256 + threadIdx.x;
  int n = blockIdx.y;
  if (k < K) Bt[(size_t)n * K + k] = (_Float16)W[(size_t)k * Nc + n];
}

// ---------------- pos embedding: pe = cp @ pos_w + pos_b ----------------
__global__ void pos_embed(const float* __restrict__ cp, const float* __restrict__ pw,
                          const float* __restrict__ pb, float* __restrict__ pe) {
  int row = blockIdx.x;
  int col = threadIdx.x;
  float c0 = cp[(size_t)row * 2 + 0];
  float c1 = cp[(size_t)row * 2 + 1];
  pe[(size_t)row * DIMC + col] = c0 * pw[col] + c1 * pw[DIMC + col] + pb[col];
}

// ---------------- layernorm over 256, one wave per row ----------------
// out = LN(A [+ pre]) * g + b [+ post]
__global__ __launch_bounds__(256)
void layernorm(const float* __restrict__ A, const float* __restrict__ pre,
               const float* __restrict__ post, const float* __restrict__ g,
               const float* __restrict__ b, float* __restrict__ out, int rows) {
  int w = threadIdx.x >> 5, lane = threadIdx.x & 31;
  int row = blockIdx.x * 8 + w;
  if (row >= rows) return;
  size_t rb = (size_t)row * DIMC;
  float vals[8];
  {
    const float4* p = (const float4*)(A + rb + lane * 8);
    float4 f0 = p[0], f1 = p[1];
    vals[0] = f0.x; vals[1] = f0.y; vals[2] = f0.z; vals[3] = f0.w;
    vals[4] = f1.x; vals[5] = f1.y; vals[6] = f1.z; vals[7] = f1.w;
  }
  if (pre) {
    const float4* p = (const float4*)(pre + rb + lane * 8);
    float4 f0 = p[0], f1 = p[1];
    vals[0] += f0.x; vals[1] += f0.y; vals[2] += f0.z; vals[3] += f0.w;
    vals[4] += f1.x; vals[5] += f1.y; vals[6] += f1.z; vals[7] += f1.w;
  }
  float s = 0.f, sq = 0.f;
#pragma unroll
  for (int j = 0; j < 8; ++j) { s += vals[j]; sq += vals[j] * vals[j]; }
#pragma unroll
  for (int msk = 16; msk >= 1; msk >>= 1) {
    s  += __shfl_xor(s,  msk, 32);
    sq += __shfl_xor(sq, msk, 32);
  }
  float mean = s * (1.0f / DIMC);
  float var  = sq * (1.0f / DIMC) - mean * mean;
  float rsd  = rsqrtf(var + 1e-5f);
#pragma unroll
  for (int j = 0; j < 8; ++j) {
    int col = lane * 8 + j;
    float o = (vals[j] - mean) * rsd * g[col] + b[col];
    if (post) o += post[rb + col];
    out[rb + col] = o;
  }
}

// ---------------- WMMA GEMM: C = A(MxK,f32) @ Bt^T (Bt = [Nc][K] f16) ----------------
// block 256 threads = 8 waves; tile 64x64, BK=32, double-buffered LDS;
// wave -> 16(M) x 32(N) strip (2 WMMAs per K-step).
// B tiles staged by the Tensor Data Mover (wave 0 issues, TENSORcnt fenced);
// fallback: vectorized manual staging. Requires K % 32 == 0, Nc % 8 == 0.
template <int DO_GELU>
__global__ __launch_bounds__(256)
void gemm_f16(const float* __restrict__ A, const _Float16* __restrict__ Bt,
              const float* __restrict__ bias, const float* __restrict__ resid,
              float* __restrict__ C, int M, int Nc, int K) {
  __shared__ __align__(16) _Float16 lA[2][64][40];  // [buf][m][k]
  __shared__ __align__(16) _Float16 lB[2][64][40];  // [buf][n][k]
  const int t    = threadIdx.x;
  const int m0   = blockIdx.y * 64;
  const int n0   = blockIdx.x * 64;
  const int w    = t >> 5, lane = t & 31;
  const int lm   = lane & 15;
  const bool hiL = lane >= 16;
  const int mw   = (w & 3) * 16;
  const int nw   = (w >> 2) * 32;

  const int ar = t >> 2, ac = (t & 3) * 8;   // A loader: 64 rows x (4 thr * 8 f32)
  const bool okA = (m0 + ar) < M;
  const float* pA = A + (size_t)(m0 + ar) * K + ac;

  const int T = K >> 5;
  const float4 z4 = make_float4(0.f, 0.f, 0.f, 0.f);
  float4 fa0 = z4, fa1 = z4;

#ifdef USE_TDM
  const bool issuer = (t < 32);  // wave 0 drives the TDM
  const _Float16* BtTile = Bt + (size_t)n0 * K;
  int rem = Nc - n0; if (rem > 64) rem = 64;
  if (issuer) tdm_load_tileB(BtTile, (unsigned)(uintptr_t)&lB[0][0][0], rem, K);
#else
  const int brow = t >> 2, bkc = (t & 3) * 8;  // B loader: 64 rows x (4 thr * 8 f16)
  const bool okB = (n0 + brow) < Nc;
  const _Float16* pB = Bt + (size_t)(n0 + brow) * K + bkc;
  v8h hb = {};
  if (okB) hb = *(const v8h*)pB;
#endif

  // prologue: tile 0 -> LDS buf 0
  if (okA) { const float4* p = (const float4*)pA; fa0 = p[0]; fa1 = p[1]; }
  *(v8h*)&lA[0][ar][ac] = cvt8(fa0, fa1);
#ifdef USE_TDM
  if (issuer) __builtin_amdgcn_s_wait_tensorcnt(0);
#else
  *(v8h*)&lB[0][brow][bkc] = hb;
#endif

  v8f acc0 = {}; v8f acc1 = {};
#pragma unroll 2
  for (int kt = 0; kt < T; ++kt) {
    __syncthreads();
    const int cur = kt & 1;
    // prefetch next tile (A -> regs, B -> TDM into other LDS buffer)
    if (kt + 1 < T) {
      fa0 = z4; fa1 = z4;
      if (okA) { const float4* p = (const float4*)(pA + (kt + 1) * 32); fa0 = p[0]; fa1 = p[1]; }
#ifdef USE_TDM
      if (issuer)
        tdm_load_tileB(BtTile + (kt + 1) * 32,
                       (unsigned)(uintptr_t)&lB[cur ^ 1][0][0], rem, K);
#else
      hb = v8h{};
      if (okB) hb = *(const v8h*)(pB + (kt + 1) * 32);
#endif
    }
    // compute from current buffer
    {
      int kb = hiL ? 16 : 0;
      v16h af = mk16(*(const v8h*)&lA[cur][mw + lm][kb],
                     *(const v8h*)&lA[cur][mw + lm][kb + 8]);
      int kb2 = hiL ? 16 : 0;
      v16h bf0 = mk16(*(const v8h*)&lB[cur][nw + lm][kb2],
                      *(const v8h*)&lB[cur][nw + lm][kb2 + 8]);
      v16h bf1 = mk16(*(const v8h*)&lB[cur][nw + 16 + lm][kb2],
                      *(const v8h*)&lB[cur][nw + 16 + lm][kb2 + 8]);
      acc0 = __builtin_amdgcn_wmma_f32_16x16x32_f16(false, af, false, bf0, (short)0, acc0, false, false);
      acc1 = __builtin_amdgcn_wmma_f32_16x16x32_f16(false, af, false, bf1, (short)0, acc1, false, false);
    }
    // publish next tile
    if (kt + 1 < T) {
      const int nxt = cur ^ 1;
      *(v8h*)&lA[nxt][ar][ac] = cvt8(fa0, fa1);
#ifdef USE_TDM
      if (issuer) __builtin_amdgcn_s_wait_tensorcnt(0);
#else
      *(v8h*)&lB[nxt][brow][bkc] = hb;
#endif
    }
  }

#pragma unroll
  for (int v = 0; v < 8; ++v) {
    int row = m0 + mw + v + (hiL ? 8 : 0);
    if (row >= M) continue;
    size_t rb = (size_t)row * Nc;
    int c0 = n0 + nw + lm;
    if (c0 < Nc) {
      float val = acc0[v];
      if (bias)    val += bias[c0];
      if (DO_GELU) val = gelu_f(val);
      if (resid)   val += resid[rb + c0];
      C[rb + c0] = val;
    }
    int c1 = c0 + 16;
    if (c1 < Nc) {
      float val = acc1[v];
      if (bias)    val += bias[c1];
      if (DO_GELU) val = gelu_f(val);
      if (resid)   val += resid[rb + c1];
      C[rb + c1] = val;
    }
  }
}

// ---------------- flash self-attention over qkv [N*LQ, 768] ----------------
// grid: (ceil(LQ/128), N*HEADS); 8 waves/block, 16 query rows per wave.
// K/V chunks (32 keys x 32 d) staged cooperatively into LDS once per block.
// Softmax uses one tile-wide running max (exact by shift-invariance) and
// row sums via an extra WMMA against an all-ones B fragment.
__global__ __launch_bounds__(256)
void attn_flash(const float* __restrict__ qkv, float* __restrict__ o) {
  __shared__ __align__(16) _Float16 lK[32][40];       // [key][d]
  __shared__ __align__(16) _Float16 lV[32][40];       // [d][key]  (transposed)
  __shared__ __align__(16) _Float16 lP[8][16][40];    // per-wave P reshape patch
  const int t    = threadIdx.x;
  const int w    = t >> 5, lane = t & 31;
  const int lm   = lane & 15;
  const bool hiL = lane >= 16;
  const int nh   = blockIdx.y;
  const int n    = nh >> 3, h = nh & 7;
  const int q0   = blockIdx.x * 128 + w * 16;
  const float* base = qkv + (size_t)n * LQ * 768;

  const int lkey = t >> 3;        // 0..31 : staged key row
  const int ldg  = (t & 7) * 4;   // 0..28 : staged d group

  // Q fragment (attention scale folded), vectorized loads
  v16h qf;
  {
    int qr = q0 + lm;
    int kb = hiL ? 16 : 0;
    const float4 z4 = make_float4(0.f, 0.f, 0.f, 0.f);
    float4 a0 = z4, a1 = z4, b0 = z4, b1 = z4;
    if (qr < LQ) {
      const float* qp = base + (size_t)qr * 768 + h * DHEAD;
      const float4* p0 = (const float4*)(qp + kb);
      a0 = p0[0]; a1 = p0[1];
      const float4* p1 = (const float4*)(qp + kb + 8);
      b0 = p1[0]; b1 = p1[1];
    }
    qf = mk16(cvt8s(a0, a1, ATT_SCALE), cvt8s(b0, b1, ATT_SCALE));
  }

  // all-ones B fragment for row sums (no loads)
  v16h ones16;
#pragma unroll
  for (int j = 0; j < 16; ++j) ones16[j] = (_Float16)1.0f;

  v8f O0 = {}; v8f O1 = {};
  float mrun = -1e30f;
  float lrun[8];
#pragma unroll
  for (int v = 0; v < 8; ++v) lrun[v] = 0.0f;

  for (int j0 = 0; j0 < LQ; j0 += 32) {
    __syncthreads();  // previous chunk fully consumed
    {  // cooperative stage of K (row-major) and V (transposed) chunks
      int key = j0 + lkey;
      float4 fk = make_float4(0.f, 0.f, 0.f, 0.f), fv = fk;
      if (key < LQ) {
        const float* kp = base + (size_t)key * 768 + 256 + h * DHEAD + ldg;
        fk = *(const float4*)kp;
        const float* vp = base + (size_t)key * 768 + 512 + h * DHEAD + ldg;
        fv = *(const float4*)vp;
      }
      v4h hk; hk[0] = (_Float16)fk.x; hk[1] = (_Float16)fk.y;
              hk[2] = (_Float16)fk.z; hk[3] = (_Float16)fk.w;
      *(v4h*)&lK[lkey][ldg] = hk;
      lV[ldg + 0][lkey] = (_Float16)fv.x;
      lV[ldg + 1][lkey] = (_Float16)fv.y;
      lV[ldg + 2][lkey] = (_Float16)fv.z;
      lV[ldg + 3][lkey] = (_Float16)fv.w;
    }
    __syncthreads();  // chunk ready

    int kb2 = hiL ? 16 : 0;
    v16h kf0 = mk16(*(const v8h*)&lK[lm][kb2],      *(const v8h*)&lK[lm][kb2 + 8]);
    v16h kf1 = mk16(*(const v8h*)&lK[lm + 16][kb2], *(const v8h*)&lK[lm + 16][kb2 + 8]);
    v8f z = {};
    v8f S0 = __builtin_amdgcn_wmma_f32_16x16x32_f16(false, qf, false, kf0, (short)0, z, false, false);
    v8f S1 = __builtin_amdgcn_wmma_f32_16x16x32_f16(false, qf, false, kf1, (short)0, z, false, false);

    // tile-wide max: OOB keys have S == 0 exactly (K staged as zeros)
    float mx = S0[0];
#pragma unroll
    for (int v = 0; v < 8; ++v) { mx = fmaxf(mx, S0[v]); mx = fmaxf(mx, S1[v]); }
#pragma unroll
    for (int msk = 16; msk >= 1; msk >>= 1) mx = fmaxf(mx, __shfl_xor(mx, msk, 32));
    float mnew  = fmaxf(mrun, mx);
    float alpha = __expf(mrun - mnew);
    mrun = mnew;

    float mask0 = ((j0 + lm) < LQ)      ? 1.0f : 0.0f;
    float mask1 = ((j0 + 16 + lm) < LQ) ? 1.0f : 0.0f;
    // P: C-layout -> A-fragment reshape via per-wave LDS patch
#pragma unroll
    for (int v = 0; v < 8; ++v) {
      int m = v + (hiL ? 8 : 0);
      lP[w][m][lm]      = (_Float16)(mask0 * __expf(S0[v] - mnew));
      lP[w][m][lm + 16] = (_Float16)(mask1 * __expf(S1[v] - mnew));
    }
    v16h pf;
    {
      int kb = hiL ? 16 : 0;
      pf = mk16(*(const v8h*)&lP[w][lm][kb], *(const v8h*)&lP[w][lm][kb + 8]);
    }
    // row sums via matrix pipe: R[m][*] = sum_k P[m][k]
    v8f R = __builtin_amdgcn_wmma_f32_16x16x32_f16(false, pf, false, ones16, (short)0, z, false, false);

    v16h vf0 = mk16(*(const v8h*)&lV[lm][kb2],      *(const v8h*)&lV[lm][kb2 + 8]);
    v16h vf1 = mk16(*(const v8h*)&lV[lm + 16][kb2], *(const v8h*)&lV[lm + 16][kb2 + 8]);
#pragma unroll
    for (int v = 0; v < 8; ++v) {
      lrun[v] = lrun[v] * alpha + R[v];
      O0[v] *= alpha;
      O1[v] *= alpha;
    }
    O0 = __builtin_amdgcn_wmma_f32_16x16x32_f16(false, pf, false, vf0, (short)0, O0, false, false);
    O1 = __builtin_amdgcn_wmma_f32_16x16x32_f16(false, pf, false, vf1, (short)0, O1, false, false);
  }

#pragma unroll
  for (int v = 0; v < 8; ++v) {
    int row = q0 + v + (hiL ? 8 : 0);
    if (row < LQ) {
      float inv = 1.0f / lrun[v];
      size_t ob = ((size_t)n * LQ + row) * DIMC + h * DHEAD;
      o[ob + lm]      = O0[v] * inv;
      o[ob + lm + 16] = O1[v] * inv;
    }
  }
}

// ---------------- attention-weight softmax over 20 per (n,q,h) ----------------
__global__ void softmax_aw(float* __restrict__ aw) {
  int idx = blockIdx.x * blockDim.x + threadIdx.x;
  if (idx >= QROWS * HEADS) return;
  float* p = aw + (size_t)(idx / HEADS) * (HEADS * LVLS * NPTS) + (idx % HEADS) * (LVLS * NPTS);
  float v[LVLS * NPTS];
  float mx = -1e30f;
#pragma unroll
  for (int j = 0; j < LVLS * NPTS; ++j) { v[j] = p[j]; mx = fmaxf(mx, v[j]); }
  float s = 0.f;
#pragma unroll
  for (int j = 0; j < LVLS * NPTS; ++j) { v[j] = __expf(v[j] - mx); s += v[j]; }
  float inv = 1.0f / s;
#pragma unroll
  for (int j = 0; j < LVLS * NPTS; ++j) p[j] = v[j] * inv;
}

// ---------------- deformable sampling: one wave per (n,q,h); lane = channel d ----------------
__global__ __launch_bounds__(256)
void deform_sample(const float* __restrict__ value, const float* __restrict__ cp,
                   const float* __restrict__ off, const float* __restrict__ aw,
                   float* __restrict__ out) {
  const int w = threadIdx.x >> 5;
  const int d = threadIdx.x & 31;
  int id = blockIdx.x * 8 + w;  // (n*LQ+q)*HEADS + h
  if (id >= QROWS * HEADS) return;
  int h  = id % HEADS;
  int nq = id / HEADS;  // n*LQ + q
  int n  = nq / LQ;
  const int HL[LVLS] = {128, 64, 32, 16, 8};
  const int WL[LVLS] = {128, 64, 32, 16, 8};
  const int ST[LVLS] = {0, 16384, 20480, 21504, 21760};
  float cx = cp[(size_t)nq * 2 + 0];
  float cy = cp[(size_t)nq * 2 + 1];
  const float* offp = off + (size_t)nq * (HEADS * LVLS * NPTS * 2) + h * (LVLS * NPTS * 2);
  const float* awp  = aw  + (size_t)nq * (HEADS * LVLS * NPTS)     + h * (LVLS * NPTS);
  const float* vb   = value + ((size_t)n * LINR) * DIMC + h * DHEAD + d;
  float acc = 0.0f;
#pragma unroll
  for (int l = 0; l < LVLS; ++l) {
    int Hl = HL[l], Wl = WL[l], s = ST[l];
#pragma unroll
    for (int p = 0; p < NPTS; ++p) {
      float ox = offp[(l * NPTS + p) * 2 + 0];
      float oy = offp[(l * NPTS + p) * 2 + 1];
      float a  = awp[l * NPTS + p];
      float gx = cx * (float)Wl + ox - 0.5f;
      float gy = cy * (float)Hl + oy - 0.5f;
      float x0f = floorf(gx), y0f = floorf(gy);
      float lx = gx - x0f, ly = gy - y0f;
      int x0 = (int)x0f, y0 = (int)y0f;
      float s00 = 0.f, s01 = 0.f, s10 = 0.f, s11 = 0.f;
      if ((unsigned)y0 < (unsigned)Hl) {
        if ((unsigned)x0       < (unsigned)Wl) s00 = vb[(size_t)(s + y0 * Wl + x0) * DIMC];
        if ((unsigned)(x0 + 1) < (unsigned)Wl) s01 = vb[(size_t)(s + y0 * Wl + x0 + 1) * DIMC];
      }
      if ((unsigned)(y0 + 1) < (unsigned)Hl) {
        if ((unsigned)x0       < (unsigned)Wl) s10 = vb[(size_t)(s + (y0 + 1) * Wl + x0) * DIMC];
        if ((unsigned)(x0 + 1) < (unsigned)Wl) s11 = vb[(size_t)(s + (y0 + 1) * Wl + x0 + 1) * DIMC];
      }
      float samp = s00 * (1.f - ly) * (1.f - lx) + s01 * (1.f - ly) * lx
                 + s10 * ly * (1.f - lx)         + s11 * ly * lx;
      acc += a * samp;
    }
  }
  out[(size_t)nq * DIMC + h * DHEAD + d] = acc;
}

// ---------------- host orchestration ----------------
extern "C" void kernel_launch(void* const* d_in, const int* in_sizes, int n_in,
                              void* d_out, int out_size, void* d_ws, size_t ws_size,
                              hipStream_t stream) {
  (void)in_sizes; (void)n_in; (void)out_size; (void)ws_size;
  const float* x     = (const float*)d_in[0];
  const float* src   = (const float*)d_in[1];
  const float* cp    = (const float*)d_in[2];
  const float* pos_w = (const float*)d_in[5];
  const float* pos_b = (const float*)d_in[6];
  const float* ln1_g = (const float*)d_in[7];
  const float* ln1_b = (const float*)d_in[8];
  const float* qkv_w = (const float*)d_in[9];
  const float* out_w = (const float*)d_in[10];
  const float* out_b = (const float*)d_in[11];
  const float* ln2_g = (const float*)d_in[12];
  const float* ln2_b = (const float*)d_in[13];
  const float* off_w = (const float*)d_in[14];
  const float* off_b = (const float*)d_in[15];
  const float* aw_w  = (const float*)d_in[16];
  const float* aw_b  = (const float*)d_in[17];
  const float* val_w = (const float*)d_in[18];
  const float* val_b = (const float*)d_in[19];
  const float* op_w  = (const float*)d_in[20];
  const float* op_b  = (const float*)d_in[21];
  const float* ln3_g = (const float*)d_in[22];
  const float* ln3_b = (const float*)d_in[23];
  const float* ff_w1 = (const float*)d_in[24];
  const float* ff_b1 = (const float*)d_in[25];
  const float* ff_w2 = (const float*)d_in[26];
  const float* ff_b2 = (const float*)d_in[27];
  float* xo = (float*)d_out;  // running residual stream x

  // workspace layout (floats)
  float* ws = (float*)d_ws;
  size_t o = 0;
  float* pe    = ws + o; o += (size_t)QROWS * DIMC;
  float* query = ws + o; o += (size_t)QROWS * DIMC;
  float* xn    = ws + o; o += (size_t)QROWS * DIMC;
  float* obuf  = ws + o; o += (size_t)QROWS * DIMC;
  float* offb  = ws + o; o += (size_t)QROWS * 320;
  float* awb   = ws + o; o += (size_t)QROWS * 160;
  float* h1    = ws + o; o += (size_t)QROWS * MLPD;
  float* value = ws + o; o += (size_t)SROWS * DIMC;
  float* srcn  = ws + o; o += (size_t)SROWS * DIMC;
  float* qkvb  = srcn;  // qkv (4000x768) aliases srcn region; lifetimes are disjoint

  // f16 transposed weights: Bt[Nc][K]
  _Float16* wh = (_Float16*)(ws + o);
  size_t ho = 0;
  _Float16 *qkvT[2], *outT[2], *valT[2], *offT[2], *awT[2], *opT[2], *ff1T[2], *ff2T[2];
  for (int i = 0; i < 2; ++i) {
    qkvT[i] = wh + ho; ho += (size_t)768 * DIMC;
    outT[i] = wh + ho; ho += (size_t)DIMC * DIMC;
    valT[i] = wh + ho; ho += (size_t)DIMC * DIMC;
    offT[i] = wh + ho; ho += (size_t)320 * DIMC;
    awT[i]  = wh + ho; ho += (size_t)160 * DIMC;
    opT[i]  = wh + ho; ho += (size_t)DIMC * DIMC;
    ff1T[i] = wh + ho; ho += (size_t)MLPD * DIMC;
    ff2T[i] = wh + ho; ho += (size_t)DIMC * MLPD;
  }

  (void)hipMemcpyAsync(xo, x, (size_t)QROWS * DIMC * sizeof(float),
                       hipMemcpyDeviceToDevice, stream);
  pos_embed<<<QROWS, DIMC, 0, stream>>>(cp, pos_w, pos_b, pe);

  // transpose+convert all weights once per launch
  for (int i = 0; i < 2; ++i) {
    wtrans<<<dim3(1, 768), 256, 0, stream>>>(qkv_w + (size_t)i * DIMC * 768, qkvT[i], DIMC, 768);
    wtrans<<<dim3(1, 256), 256, 0, stream>>>(out_w + (size_t)i * DIMC * DIMC, outT[i], DIMC, DIMC);
    wtrans<<<dim3(1, 256), 256, 0, stream>>>(val_w + (size_t)i * DIMC * DIMC, valT[i], DIMC, DIMC);
    wtrans<<<dim3(1, 320), 256, 0, stream>>>(off_w + (size_t)i * DIMC * 320, offT[i], DIMC, 320);
    wtrans<<<dim3(1, 160), 256, 0, stream>>>(aw_w  + (size_t)i * DIMC * 160, awT[i],  DIMC, 160);
    wtrans<<<dim3(1, 256), 256, 0, stream>>>(op_w  + (size_t)i * DIMC * DIMC, opT[i], DIMC, DIMC);
    wtrans<<<dim3(1, 512), 256, 0, stream>>>(ff_w1 + (size_t)i * DIMC * MLPD, ff1T[i], DIMC, MLPD);
    wtrans<<<dim3(2, 256), 256, 0, stream>>>(ff_w2 + (size_t)i * MLPD * DIMC, ff2T[i], MLPD, DIMC);
  }

  dim3 blk(256);
  const dim3 gM4000_256(4, (QROWS + 63) / 64);  // Nc=256, M=4000

  for (int i = 0; i < 2; ++i) {
    const float* l1g = ln1_g + i * DIMC, *l1b = ln1_b + i * DIMC;
    const float* l2g = ln2_g + i * DIMC, *l2b = ln2_b + i * DIMC;
    const float* l3g = ln3_g + i * DIMC, *l3b = ln3_b + i * DIMC;

    // --- self attention ---
    layernorm<<<QROWS / 8, blk, 0, stream>>>(xo, pe, nullptr, l1g, l1b, xn, QROWS);
    gemm_f16<0><<<dim3(12, (QROWS + 63) / 64), blk, 0, stream>>>(
        xn, qkvT[i], nullptr, nullptr, qkvb, QROWS, 768, DIMC);
    attn_flash<<<dim3((LQ + 127) / 128, NB * HEADS), blk, 0, stream>>>(qkvb, obuf);
    gemm_f16<0><<<gM4000_256, blk, 0, stream>>>(
        obuf, outT[i], out_b + i * DIMC, xo, xo, QROWS, DIMC, DIMC);

    // --- deformable cross attention ---
    layernorm<<<QROWS / 8, blk, 0, stream>>>(xo, nullptr, pe, l2g, l2b, query, QROWS);
    layernorm<<<SROWS / 8, blk, 0, stream>>>(src, nullptr, nullptr, l2g, l2b, srcn, SROWS);
    gemm_f16<0><<<dim3(4, SROWS / 64), blk, 0, stream>>>(
        srcn, valT[i], val_b + i * DIMC, nullptr, value, SROWS, DIMC, DIMC);
    gemm_f16<0><<<dim3(5, (QROWS + 63) / 64), blk, 0, stream>>>(
        query, offT[i], off_b + i * 320, nullptr, offb, QROWS, 320, DIMC);
    gemm_f16<0><<<dim3(3, (QROWS + 63) / 64), blk, 0, stream>>>(
        query, awT[i], aw_b + i * 160, nullptr, awb, QROWS, 160, DIMC);
    softmax_aw<<<(QROWS * HEADS + 255) / 256, blk, 0, stream>>>(awb);
    deform_sample<<<(QROWS * HEADS) / 8, blk, 0, stream>>>(value, cp, offb, awb, obuf);
    gemm_f16<0><<<gM4000_256, blk, 0, stream>>>(
        obuf, opT[i], op_b + i * DIMC, xo, xo, QROWS, DIMC, DIMC);

    // --- feed forward ---
    layernorm<<<QROWS / 8, blk, 0, stream>>>(xo, nullptr, nullptr, l3g, l3b, xn, QROWS);
    gemm_f16<1><<<dim3(8, (QROWS + 63) / 64), blk, 0, stream>>>(
        xn, ff1T[i], ff_b1 + i * MLPD, nullptr, h1, QROWS, MLPD, DIMC);
    gemm_f16<0><<<gM4000_256, blk, 0, stream>>>(
        h1, ff2T[i], ff_b2 + i * DIMC, xo, xo, QROWS, DIMC, MLPD);
  }
}
